// DSQGAttentionV5Cuda_61272003444840
// MI455X (gfx1250) — matrix operand order repaired and checked by reading, not compiled, not run on hardware
//
#include <hip/hip_runtime.h>
#include <hip/hip_bf16.h>

#if defined(__has_builtin)
#  if __has_builtin(__builtin_amdgcn_global_load_async_to_lds_b128)
#    define HAVE_ASYNC_LDS 1
#  endif
#endif
#if defined(HAVE_ASYNC_LDS)
#  if __has_builtin(__builtin_amdgcn_global_load_async_to_lds_b32)
#    define HAVE_ASYNC_TZ 1
#  endif
#endif

typedef __attribute__((ext_vector_type(16))) _Float16 v16h;
typedef __attribute__((ext_vector_type(8)))  float    v8f;
typedef __attribute__((ext_vector_type(4)))  float    v4f;
typedef __attribute__((ext_vector_type(4)))  int      v4i;

typedef v4i __attribute__((address_space(1)))* gv4i_p;  // global (64-bit ptr)
typedef v4i __attribute__((address_space(3)))* lv4i_p;  // LDS (32-bit ptr)
typedef int __attribute__((address_space(1)))* gi_p;
typedef int __attribute__((address_space(3)))* li_p;

#define NB44 44
#define NB11 11
#define HD   64
#define HCNT 16
#define NSEQ 1024
#define BCNT 2

__device__ __forceinline__ float redmax16(float v) {
  v = fmaxf(v, __shfl_xor(v, 1, 32));
  v = fmaxf(v, __shfl_xor(v, 2, 32));
  v = fmaxf(v, __shfl_xor(v, 4, 32));
  v = fmaxf(v, __shfl_xor(v, 8, 32));
  return v;
}
__device__ __forceinline__ float redsum16(float v) {
  v += __shfl_xor(v, 1, 32);
  v += __shfl_xor(v, 2, 32);
  v += __shfl_xor(v, 4, 32);
  v += __shfl_xor(v, 8, 32);
  return v;
}

// ---------------- Prologue 1: ty/tz probe-phase tables ----------------
__global__ void probe_phase_kernel(const float* __restrict__ q,
                                   const float* __restrict__ k,
                                   const float* __restrict__ phase_base,
                                   const float* __restrict__ phase_gain,
                                   const float* __restrict__ qprobe,
                                   const float* __restrict__ kprobe,
                                   float* __restrict__ ty,
                                   float* __restrict__ tz, int total) {
  int idx = blockIdx.x * blockDim.x + threadIdx.x;
  if (idx >= total) return;
  int h = (idx / NSEQ) % HCNT;
  const float sc = 0.125f;  // 64^-0.5
  float y0 = 0.f, y1 = 0.f, z0 = 0.f, z1 = 0.f;
  const float* qp = q + (size_t)idx * HD;
  const float* kp = k + (size_t)idx * HD;
#pragma unroll 8
  for (int d = 0; d < HD; ++d) {
    y0 += qp[d] * qprobe[d];
    y1 += qp[d] * qprobe[HD + d];
    z0 += kp[d] * kprobe[d];
    z1 += kp[d] * kprobe[HD + d];
  }
  y0 *= sc; y1 *= sc; z0 *= sc; z1 *= sc;
#pragma unroll
  for (int g = 0; g < NB11; ++g) {
    int base = (g * HCNT + h) * 2;
    float pb0 = phase_base[base], pb1 = phase_base[base + 1];
    float pg0 = phase_gain[base], pg1 = phase_gain[base + 1];
    ty[(size_t)idx * NB11 + g] = y0 * (pg0 * __cosf(pb0)) + y1 * (pg1 * __cosf(pb1));
    tz[(size_t)idx * NB11 + g] = z0 * (pg0 * __sinf(pb0)) + z1 * (pg1 * __sinf(pb1));
  }
}

// ---------------- Prologue 2: qr content-bias table ----------------
__global__ void qr_kernel(const float* __restrict__ q,
                          const float* __restrict__ scale_embed,
                          float* __restrict__ qr, int total) {
  int idx = blockIdx.x * blockDim.x + threadIdx.x;
  if (idx >= total) return;
  int row = idx / NB44;
  int g   = idx % NB44;
  const float* qp = q + (size_t)row * HD;
  const float* sp = scale_embed + (size_t)g * HD;
  float acc = 0.f;
#pragma unroll 8
  for (int d = 0; d < HD; ++d) acc += qp[d] * sp[d];
  qr[idx] = acc * 0.125f;
}

// ---------------- Main: flash attention with WMMA ----------------
__global__ __launch_bounds__(128, 1) void fa_wmma_kernel(
    const float* __restrict__ q, const float* __restrict__ k,
    const float* __restrict__ v, const float* __restrict__ pos_bias,
    const float* __restrict__ qr_ws, const float* __restrict__ ty_ws,
    const float* __restrict__ tz_ws, float* __restrict__ out) {
  __shared__ unsigned short btab[NSEQ];        // packed b44 | b11<<8 per distance
  __shared__ float    qrpbL[4][16 * NB44];     // qr + pos_bias, per wave rows
  __shared__ float    tyL[4][16 * NB11];
  __shared__ float    tzL[4][32 * NB11];       // tz slice for current j-tile
  __shared__ _Float16 Plds[4][16 * 32];
  __shared__ float    Vlds[4][32 * HD];        // V tile staged as f32 (async)

  const int lane = threadIdx.x & 31;
  const int wave = threadIdx.x >> 5;
  const int bh   = blockIdx.x >> 4;                  // 0..31  (B*H)
  const int h    = bh & (HCNT - 1);
  const int rb   = ((blockIdx.x & 15) << 2) | wave;  // 0..63 row blocks
  const int i0   = rb << 4;
  const int m    = lane & 15;
  const int half = lane >> 4;

  // ---- one-time bucket table (replaces per-element log2/floor) ----
  for (int d = threadIdx.x; d < NSEQ; d += 128) {
    float df  = (float)d;
    int   b44 = min((int)floorf(4.0f * __log2f(df + 1.0f)), NB44 - 1);
    int   b11 = (d < 1) ? 0 : min((int)floorf(__log2f(df)) + 1, NB11 - 1);
    btab[d] = (unsigned short)(b44 | (b11 << 8));
  }

  // ---- stage per-wave tables (pos_bias folded into qr rows) ----
  const float* qrRow = qr_ws + ((size_t)bh * NSEQ + i0) * NB44;
  for (int t = lane; t < 16 * NB44; t += 32) {
    int g = t % NB44;
    qrpbL[wave][t] = qrRow[t] + pos_bias[g * HCNT + h];
  }
  const float* tyRow = ty_ws + ((size_t)bh * NSEQ + i0) * NB11;
  for (int t = lane; t < 16 * NB11; t += 32) tyL[wave][t] = tyRow[t];
  __syncthreads();

  // ---- Q fragments (A layout, 1/sqrt(HD) folded in) ----
  const float* qrow = q + ((size_t)bh * NSEQ + i0 + m) * HD;
  v16h a0, a1;
#pragma unroll
  for (int e = 0; e < 8; ++e) {
    a0[e]     = (_Float16)(qrow[16 * half + e] * 0.125f);
    a0[e + 8] = (_Float16)(qrow[16 * half + 8 + e] * 0.125f);
    a1[e]     = (_Float16)(qrow[32 + 16 * half + e] * 0.125f);
    a1[e + 8] = (_Float16)(qrow[32 + 16 * half + 8 + e] * 0.125f);
  }

  v8f o0 = {}, o1 = {}, o2 = {}, o3 = {};
  float rowM[8], rowL[8];
#pragma unroll
  for (int r = 0; r < 8; ++r) { rowM[r] = -3.0e38f; rowL[r] = 0.f; }

  const float* tzp  = tz_ws + (size_t)bh * NSEQ * NB11;
  const int    jend = i0 + 15;

  for (int j0 = 0; j0 <= jend; j0 += 32) {
    // ---- stage tz slice FIRST via async b32 (completes before V's b128s) ----
#if defined(HAVE_ASYNC_TZ)
    {
      const float* tzt = tzp + (size_t)j0 * NB11 + lane;
      gi_p gp = __builtin_bit_cast(gi_p, tzt);
      li_p lp = __builtin_bit_cast(li_p, (unsigned int)(size_t)(&tzL[wave][lane]));
#pragma unroll
      for (int it = 0; it < NB11; ++it)
        __builtin_amdgcn_global_load_async_to_lds_b32(gp + 32 * it, lp + 32 * it, 0, 0);
    }
#else
    {
      const float* tzt = tzp + (size_t)j0 * NB11;
#pragma unroll
      for (int it = 0; it < NB11; ++it)
        tzL[wave][lane + 32 * it] = tzt[lane + 32 * it];
    }
#endif

    // ---- stage V tile: CDNA5 async global->LDS (overlaps S WMMAs) ----
#if defined(HAVE_ASYNC_LDS)
    {
      const float* vr = v + ((size_t)bh * NSEQ + j0 + lane) * HD;
      gv4i_p gp = __builtin_bit_cast(gv4i_p, vr);
      lv4i_p lp = __builtin_bit_cast(lv4i_p,
                      (unsigned int)(size_t)(&Vlds[wave][lane * HD]));
#pragma unroll
      for (int t = 0; t < 16; ++t)
        __builtin_amdgcn_global_load_async_to_lds_b128(gp + t, lp + t, 0, 0);
    }
#else
    {
      const v4f* vr4 = (const v4f*)(v + ((size_t)bh * NSEQ + j0 + lane) * HD);
      v4f*       vl4 = (v4f*)(&Vlds[wave][lane * HD]);
#pragma unroll
      for (int t = 0; t < 16; ++t) vl4[t] = vr4[t];
    }
#endif

    // ---- K fragments (B layout: 32x16 f16, column = key row) ----
    const float* kr0 = k + ((size_t)bh * NSEQ + j0 + m) * HD + 16 * half;
    const float* kr1 = kr0 + (size_t)16 * HD;
    v16h b0, b1, b2, b3;
#pragma unroll
    for (int e = 0; e < 16; ++e) {
      b0[e] = (_Float16)kr0[e];
      b1[e] = (_Float16)kr0[32 + e];
      b2[e] = (_Float16)kr1[e];
      b3[e] = (_Float16)kr1[32 + e];
    }
    v8f s0 = {}, s1 = {};
    s0 = __builtin_amdgcn_wmma_f32_16x16x32_f16(false, a0, false, b0, (short)0, s0, false, false);
    s0 = __builtin_amdgcn_wmma_f32_16x16x32_f16(false, a1, false, b1, (short)0, s0, false, false);
    s1 = __builtin_amdgcn_wmma_f32_16x16x32_f16(false, a0, false, b2, (short)0, s1, false, false);
    s1 = __builtin_amdgcn_wmma_f32_16x16x32_f16(false, a1, false, b3, (short)0, s1, false, false);

    // ---- tz asyncs (first 11 issued) are done once <=16 remain outstanding ----
#if defined(HAVE_ASYNC_TZ)
#  if __has_builtin(__builtin_amdgcn_s_wait_asynccnt)
    __builtin_amdgcn_s_wait_asynccnt(16);
#  else
    asm volatile("s_wait_asynccnt 16" ::: "memory");
#  endif
#endif

    // ---- biases via LDS tables + causal mask ----
#pragma unroll
    for (int r = 0; r < 8; ++r) {
      const int M  = r + 8 * half;
      const int i  = i0 + M;
      const int d0 = i - (j0 + m);
      if (d0 < 0) s0[r] = -1e30f;
      else {
        int bt = btab[d0];
        int b44 = bt & 0xFF, b11 = bt >> 8;
        s0[r] += qrpbL[wave][M * NB44 + b44] + tyL[wave][M * NB11 + b11]
               + tzL[wave][m * NB11 + b11];
      }
      const int d1 = d0 - 16;
      if (d1 < 0) s1[r] = -1e30f;
      else {
        int bt = btab[d1];
        int b44 = bt & 0xFF, b11 = bt >> 8;
        s1[r] += qrpbL[wave][M * NB44 + b44] + tyL[wave][M * NB11 + b11]
               + tzL[wave][(16 + m) * NB11 + b11];
      }
    }

    // ---- online softmax update; write P (f16) to wave-private LDS ----
#pragma unroll
    for (int r = 0; r < 8; ++r) {
      float t    = redmax16(fmaxf(s0[r], s1[r]));
      float mNew = fmaxf(rowM[r], t);
      float p0   = __expf(s0[r] - mNew);
      float p1   = __expf(s1[r] - mNew);
      float srow = redsum16(p0 + p1);
      float f    = __expf(rowM[r] - mNew);
      rowL[r] = rowL[r] * f + srow;
      rowM[r] = mNew;
      o0[r] *= f; o1[r] *= f; o2[r] *= f; o3[r] *= f;
      const int M = r + 8 * half;
      Plds[wave][M * 32 + m]      = (_Float16)p0;
      Plds[wave][M * 32 + 16 + m] = (_Float16)p1;
    }

    // ---- wait for async V staging just before it is consumed ----
#if defined(HAVE_ASYNC_LDS)
#  if __has_builtin(__builtin_amdgcn_s_wait_asynccnt)
    __builtin_amdgcn_s_wait_asynccnt(0);
#  else
    asm volatile("s_wait_asynccnt 0" ::: "memory");
#  endif
#endif

    // ---- P·V : A = P (16x32), B = V (32x16 per 16-col group) ----
    v16h ap;
#pragma unroll
    for (int e = 0; e < 8; ++e) {
      ap[e]     = Plds[wave][m * 32 + 16 * half + e];
      ap[e + 8] = Plds[wave][m * 32 + 16 * half + 8 + e];
    }
    v16h bv0, bv1, bv2, bv3;
#pragma unroll
    for (int e = 0; e < 16; ++e) {
      const int kk = 16 * half + e;
      bv0[e] = (_Float16)Vlds[wave][kk * HD + 0  + m];
      bv1[e] = (_Float16)Vlds[wave][kk * HD + 16 + m];
      bv2[e] = (_Float16)Vlds[wave][kk * HD + 32 + m];
      bv3[e] = (_Float16)Vlds[wave][kk * HD + 48 + m];
    }
    o0 = __builtin_amdgcn_wmma_f32_16x16x32_f16(false, ap, false, bv0, (short)0, o0, false, false);
    o1 = __builtin_amdgcn_wmma_f32_16x16x32_f16(false, ap, false, bv1, (short)0, o1, false, false);
    o2 = __builtin_amdgcn_wmma_f32_16x16x32_f16(false, ap, false, bv2, (short)0, o2, false, false);
    o3 = __builtin_amdgcn_wmma_f32_16x16x32_f16(false, ap, false, bv3, (short)0, o3, false, false);
  }

  // ---- epilogue: normalize and store ----
#pragma unroll
  for (int r = 0; r < 8; ++r) {
    const int i   = i0 + r + 8 * half;
    float     inv = 1.0f / rowL[r];
    float*    op  = out + ((size_t)bh * NSEQ + i) * HD + m;
    op[0]  = o0[r] * inv;
    op[16] = o1[r] * inv;
    op[32] = o2[r] * inv;
    op[48] = o3[r] * inv;
  }
}

extern "C" void kernel_launch(void* const* d_in, const int* in_sizes, int n_in,
                              void* d_out, int out_size, void* d_ws, size_t ws_size,
                              hipStream_t stream) {
  const float* q           = (const float*)d_in[0];
  const float* k           = (const float*)d_in[1];
  const float* v           = (const float*)d_in[2];
  const float* pos_bias    = (const float*)d_in[3];
  const float* scale_embed = (const float*)d_in[4];
  const float* phase_base  = (const float*)d_in[5];
  const float* phase_gain  = (const float*)d_in[6];
  const float* qprobes     = (const float*)d_in[7];
  const float* kprobes     = (const float*)d_in[8];
  float*       out         = (float*)d_out;

  const size_t rows = (size_t)BCNT * HCNT * NSEQ;  // 32768
  float* qr = (float*)d_ws;                         // rows*44 floats
  float* ty = qr + rows * NB44;                     // rows*11 floats
  float* tz = ty + rows * NB11;                     // rows*11 floats

  const int totalA = (int)rows;
  probe_phase_kernel<<<(totalA + 255) / 256, 256, 0, stream>>>(
      q, k, phase_base, phase_gain, qprobes, kprobes, ty, tz, totalA);

  const int totalB = (int)(rows * NB44);
  qr_kernel<<<(totalB + 255) / 256, 256, 0, stream>>>(q, scale_embed, qr, totalB);

  fa_wmma_kernel<<<BCNT * HCNT * (NSEQ / 64), 128, 0, stream>>>(
      q, k, v, pos_bias, qr, ty, tz, out);
}
